// PointNet2feat_33741263077656
// MI455X (gfx1250) — matrix operand order repaired and checked by reading, not compiled, not run on hardware
//
#include <hip/hip_runtime.h>

// ---------------------------------------------------------------------------
// PointNet++ MSG feature extractor for gfx1250 (MI455X), wave32 + WMMA.
// Pipeline: FPS -> (ball-group -> 3x[WMMA GEMM -> stats -> BN+ReLU] -> maxpool) x3
//           -> FC1(WMMA)+BN+ReLU -> FC2(WMMA)+BN+ReLU -> d_out [256,256] f32
// GEMM: N-strip register blocking (NT accumulators/wave), block-shared LDS
// staging of f16 weights, global_prefetch of the next A chunk.
// ---------------------------------------------------------------------------

typedef float    f32x4  __attribute__((ext_vector_type(4)));
typedef float    f32x8  __attribute__((ext_vector_type(8)));
typedef _Float16 f16x4  __attribute__((ext_vector_type(4)));
typedef _Float16 f16x8  __attribute__((ext_vector_type(8)));
typedef _Float16 f16x16 __attribute__((ext_vector_type(16)));

#define B_   256
#define N_   8192
#define S_   16
#define EPS_ 1e-5f

// ------------------------------- FPS ---------------------------------------
__global__ __launch_bounds__(256)
void fps_kernel(const float* __restrict__ xyz, float* __restrict__ new_xyz) {
    __shared__ float dist[N_];
    __shared__ float rv[256];
    __shared__ int   ri[256];
    __shared__ int   curFar;
    const int b = blockIdx.x;
    const int t = threadIdx.x;
    const float* px = xyz + (size_t)b * 6 * N_;
    const float* py = px + N_;
    const float* pz = px + 2 * N_;
    for (int n = t; n < N_; n += 256) dist[n] = 1e10f;
    if (t == 0) curFar = 0;
    __syncthreads();
    for (int s = 0; s < S_; ++s) {
        const int far = curFar;
        const float cx = px[far], cy = py[far], cz = pz[far];
        if (t == 0) {
            new_xyz[(b * S_ + s) * 3 + 0] = cx;
            new_xyz[(b * S_ + s) * 3 + 1] = cy;
            new_xyz[(b * S_ + s) * 3 + 2] = cz;
        }
        float bv = -1.0f; int bi = 0;
        for (int n = t; n < N_; n += 256) {
            const float dx = px[n] - cx, dy = py[n] - cy, dz = pz[n] - cz;
            const float d  = dx * dx + dy * dy + dz * dz;
            const float od = dist[n];
            const float nd = d < od ? d : od;
            dist[n] = nd;
            if (nd > bv) { bv = nd; bi = n; }      // strict > keeps lowest index
        }
        rv[t] = bv; ri[t] = bi;
        __syncthreads();
        for (int off = 128; off; off >>= 1) {
            if (t < off) {
                const float v2 = rv[t + off]; const int i2 = ri[t + off];
                if (v2 > rv[t] || (v2 == rv[t] && i2 < ri[t])) { rv[t] = v2; ri[t] = i2; }
            }
            __syncthreads();
        }
        if (t == 0) curFar = ri[0];
        __syncthreads();
    }
}

// --------------------------- Ball query + group -----------------------------
__device__ __forceinline__ void write_group_row(
    _Float16* __restrict__ g, int bs, int K, int slot, int n,
    const float* px, const float* py, const float* pz,
    const float* nx, const float* ny, const float* nz,
    float cx, float cy, float cz) {
    _Float16* row = g + ((size_t)bs * K + slot) * 32;
    row[0] = (_Float16)nx[n];
    row[1] = (_Float16)ny[n];
    row[2] = (_Float16)nz[n];
    row[3] = (_Float16)(px[n] - cx);
    row[4] = (_Float16)(py[n] - cy);
    row[5] = (_Float16)(pz[n] - cz);
#pragma unroll
    for (int c = 6; c < 32; ++c) row[c] = (_Float16)0.0f;
}

__global__ __launch_bounds__(256)
void ballgroup_kernel(const float* __restrict__ xyz,
                      const float* __restrict__ new_xyz,
                      _Float16* __restrict__ g, float r2, int K) {
    __shared__ int scan[256];
    __shared__ int firstN;
    const int bs = blockIdx.x;            // b*16 + s
    const int b  = bs >> 4;
    const int t  = threadIdx.x;
    const float* px = xyz + (size_t)b * 6 * N_;
    const float* py = px + N_;
    const float* pz = px + 2 * N_;
    const float* nx = px + 3 * N_;
    const float* ny = px + 4 * N_;
    const float* nz = px + 5 * N_;
    const float cx = new_xyz[bs * 3 + 0];
    const float cy = new_xyz[bs * 3 + 1];
    const float cz = new_xyz[bs * 3 + 2];
    const int n0 = t * 32;                // contiguous chunk -> preserves order
    int c = 0;
    for (int j = 0; j < 32; ++j) {
        const int n = n0 + j;
        const float dx = px[n] - cx, dy = py[n] - cy, dz = pz[n] - cz;
        if (dx * dx + dy * dy + dz * dz <= r2) c++;
    }
    scan[t] = c;
    __syncthreads();
    for (int off = 1; off < 256; off <<= 1) {    // Kogge-Stone inclusive scan
        const int v = (t >= off) ? scan[t - off] : 0;
        __syncthreads();
        scan[t] += v;
        __syncthreads();
    }
    const int excl  = scan[t] - c;
    const int total = scan[255];
    if (c > 0 && excl == 0) {                    // unique first-matching thread
        for (int j = 0; j < 32; ++j) {
            const int n = n0 + j;
            const float dx = px[n] - cx, dy = py[n] - cy, dz = pz[n] - cz;
            if (dx * dx + dy * dy + dz * dz <= r2) { firstN = n; break; }
        }
    }
    __syncthreads();
    int rank = 0;
    for (int j = 0; j < 32 && rank < c; ++j) {
        const int n = n0 + j;
        const float dx = px[n] - cx, dy = py[n] - cy, dz = pz[n] - cz;
        if (dx * dx + dy * dy + dz * dz <= r2) {
            const int slot = excl + rank; rank++;
            if (slot < K)
                write_group_row(g, bs, K, slot, n, px, py, pz, nx, ny, nz, cx, cy, cz);
        }
    }
    for (int slot = total + t; slot < K; slot += 256)
        write_group_row(g, bs, K, slot, firstN, px, py, pz, nx, ny, nz, cx, cy, cz);
}

// ----------------------- Zero-pad first-layer weights -----------------------
__global__ __launch_bounds__(256)
void padw_kernel(const float* __restrict__ W, float* __restrict__ Wp, int Cout) {
    const int i = blockIdx.x * 256 + threadIdx.x;
    if (i >= Cout * 32) return;
    const int o = i >> 5, c = i & 31;
    Wp[i] = (c < 6) ? W[o * 6 + c] : 0.0f;
}

// ------------------------------ WMMA GEMM -----------------------------------
// Out[m,n] = sum_k A[m,k] * W[n,k].  A: f16 row-major [M,K]; W: f32 row-major
// [N,K]; Out: f16 [M,N].
// Block = 8 waves; wave w owns M-tile (blockIdx.x*8+w); all waves share the
// N-strip blockIdx.y of NT tiles. The strip's weights are staged in LDS as
// f16 (32KB buffer, K chunked), then each wave runs: per 32-K step, one
// global A-fragment load feeding NT back-to-back v_wmma into NT accumulators.
// M/16 is always a multiple of 8 here, so no wave exits early: EXEC is
// all-ones at every WMMA and block barriers are safe.
template <int NT>
__global__ __launch_bounds__(256)
void gemm_wmma_nt(const _Float16* __restrict__ A, const float* __restrict__ W,
                  _Float16* __restrict__ Out, int M, int N, int K) {
    constexpr int KC = 16384 / (NT * 16);        // LDS K-chunk (halfs per row)
    __shared__ _Float16 wlds[NT * 16 * KC];      // 32 KB
    const int tid   = threadIdx.x;
    const int lane  = tid & 31;
    const int wave  = tid >> 5;
    const int half  = lane >> 4;                 // 0: lanes 0-15, 1: lanes 16-31
    const int l15   = lane & 15;
    const int tm    = blockIdx.x * 8 + wave;     // M-tile of this wave
    const int strip = blockIdx.y;                // N-strip of this block
    const int nbase = strip * NT * 16;
    const int arow  = tm * 16 + l15;
    const _Float16* ap = A + (size_t)arow * K + half * 8;

    f32x8 acc[NT];
#pragma unroll
    for (int t = 0; t < NT; ++t)
        acc[t] = (f32x8){0.f, 0.f, 0.f, 0.f, 0.f, 0.f, 0.f, 0.f};

    for (int k0 = 0; k0 < K; k0 += KC) {
        const int kcc = (K - k0) < KC ? (K - k0) : KC;   // multiple of 32
        // prefetch next A chunk (global_prefetch_b8, speculative)
        __builtin_prefetch(ap + k0 + KC, 0, 1);
        // ---- cooperative stage: W[nbase..nbase+NT*16)[k0..k0+kcc) -> LDS f16
        __syncthreads();
        {
            const int c4cnt = kcc >> 2;
            for (int it = tid; it < NT * 16 * c4cnt; it += 256) {
                const int n = it / c4cnt;
                const int c = (it - n * c4cnt) << 2;
                const f32x4 w4 = *(const f32x4*)(W + (size_t)(nbase + n) * K + k0 + c);
                f16x4 h4;
                h4[0] = (_Float16)w4[0]; h4[1] = (_Float16)w4[1];
                h4[2] = (_Float16)w4[2]; h4[3] = (_Float16)w4[3];
                *(f16x4*)&wlds[n * KC + c] = h4;
            }
        }
        __syncthreads();
        // ---- compute over this K chunk
        for (int kk = 0; kk < kcc; kk += 32) {
            const f16x8 alo = *(const f16x8*)(ap + k0 + kk);       // A[arow][k+h*8..]
            const f16x8 ahi = *(const f16x8*)(ap + k0 + kk + 16);  // A[arow][k+16+h*8..]
            f16x16 af;
#pragma unroll
            for (int j = 0; j < 8; ++j) { af[j] = alo[j]; af[8 + j] = ahi[j]; }
#pragma unroll
            for (int t = 0; t < NT; ++t) {
                const _Float16* bp = &wlds[(t * 16 + l15) * KC + kk + half * 8];
                const f16x8 blo = *(const f16x8*)bp;       // B[k+h*8..][col] (LDS)
                const f16x8 bhi = *(const f16x8*)(bp + 16);
                f16x16 bf;
#pragma unroll
                for (int j = 0; j < 8; ++j) { bf[j] = blo[j]; bf[8 + j] = bhi[j]; }
                acc[t] = __builtin_amdgcn_wmma_f32_16x16x32_f16(
                    false, af, false, bf, (short)0, acc[t], false, false);
            }
        }
    }
    // ---- epilogue: C layout vgpr r -> row r (lanes 0-15) / 8+r (lanes 16-31)
    const int r0 = half * 8;
#pragma unroll
    for (int t = 0; t < NT; ++t) {
        _Float16* orow = Out + (size_t)(tm * 16 + r0) * N + nbase + t * 16 + l15;
#pragma unroll
        for (int r = 0; r < 8; ++r) orow[(size_t)r * N] = (_Float16)acc[t][r];
    }
}

// ------------------- BatchNorm stats (deterministic) ------------------------
__global__ __launch_bounds__(256)
void stats_kernel(const _Float16* __restrict__ X, float* __restrict__ coef,
                  int M, int N) {
    __shared__ float sh_s[256];
    __shared__ float sh_q[256];
    const int c = blockIdx.x, t = threadIdx.x;
    float s = 0.0f, q = 0.0f;
    for (int m = t; m < M; m += 256) {
        const float v = (float)X[(size_t)m * N + c];
        s += v; q += v * v;
    }
    sh_s[t] = s; sh_q[t] = q;
    __syncthreads();
    for (int off = 128; off; off >>= 1) {
        if (t < off) { sh_s[t] += sh_s[t + off]; sh_q[t] += sh_q[t + off]; }
        __syncthreads();
    }
    if (t == 0) {
        const float mean = sh_s[0] / (float)M;
        const float var  = sh_q[0] / (float)M - mean * mean;
        const float a    = rsqrtf(var + EPS_);
        coef[c]     = a;
        coef[N + c] = -mean * a;
    }
}

// --------------------------- BN + ReLU apply --------------------------------
__global__ __launch_bounds__(256)
void bnrelu_kernel(_Float16* __restrict__ X, const float* __restrict__ coef,
                   int M, int N, float* __restrict__ outF32) {
    const size_t i = (size_t)blockIdx.x * 256 + threadIdx.x;
    if (i >= (size_t)M * N) return;
    const int c = (int)(i % N);
    float v = (float)X[i] * coef[c] + coef[N + c];
    v = v > 0.0f ? v : 0.0f;
    if (outF32) outF32[i] = v;
    else        X[i] = (_Float16)v;
}

// ------------------------------ K-max pool ----------------------------------
__global__ __launch_bounds__(256)
void pool_kernel(const _Float16* __restrict__ act, _Float16* __restrict__ fc1in,
                 int K, int C, int chanOff) {
    const int tid = blockIdx.x * 256 + threadIdx.x;
    if (tid >= B_ * S_ * C) return;
    const int c  = tid % C;
    const int bs = tid / C;
    const int s  = bs & 15;
    const int b  = bs >> 4;
    float m = -1e30f;
    const _Float16* p = act + (size_t)bs * K * C + c;
    for (int k = 0; k < K; ++k) {
        const float v = (float)p[(size_t)k * C];
        m = v > m ? v : m;
    }
    fc1in[(size_t)b * 4608 + (size_t)(chanOff + c) * 16 + s] = (_Float16)m;
}

// ---------------------------------------------------------------------------
extern "C" void kernel_launch(void* const* d_in, const int* in_sizes, int n_in,
                              void* d_out, int out_size, void* d_ws, size_t ws_size,
                              hipStream_t stream) {
    (void)out_size; (void)ws_size;
    // ---- identify inputs by flat size (robust to pytree flatten order) ----
    const float* xyz = nullptr;
    const float* fc1W = nullptr;
    int i192[2] = {-1,-1},  n192 = 0;
    int i1024[2] = {-1,-1}, n1024 = 0;
    int i8192[2] = {-1,-1}, n8192 = 0;
    int i16384[2] = {-1,-1}, n16384 = 0;
    int i2048 = -1, i384 = -1;
    for (int i = 0; i < n_in; ++i) {
        switch (in_sizes[i]) {
            case 12582912: xyz  = (const float*)d_in[i]; break;
            case 294912:   fc1W = (const float*)d_in[i]; break;
            case 192:   if (n192   < 2) i192[n192++]     = i; break;
            case 1024:  if (n1024  < 2) i1024[n1024++]   = i; break;
            case 2048:  i2048 = i; break;
            case 384:   i384  = i; break;
            case 8192:  if (n8192  < 2) i8192[n8192++]   = i; break;
            case 16384: if (n16384 < 2) i16384[n16384++] = i; break;
            default: break;
        }
    }
    // 16384 collision: scale2-L2 W (followed by a 128-vector) vs fc2_W
    // (followed by fc2_b, a 256-vector) -- holds for both flatten orders.
    int iS2L2 = -1, iFC2 = -1;
    for (int j = 0; j < n16384; ++j) {
        const int idx = i16384[j];
        if (idx + 1 < n_in && in_sizes[idx + 1] == 256) iFC2 = idx;
        else iS2L2 = idx;
    }
    if (!xyz || !fc1W || n192 < 2 || n1024 < 2 || n8192 < 2 ||
        i2048 < 0 || i384 < 0 || iS2L2 < 0 || iFC2 < 0) return;
    const float* Ws[3][3] = {
        { (const float*)d_in[i192[0]], (const float*)d_in[i1024[0]], (const float*)d_in[i1024[1]] },
        { (const float*)d_in[i192[1]], (const float*)d_in[i2048],    (const float*)d_in[i8192[0]] },
        { (const float*)d_in[i384],    (const float*)d_in[i8192[1]], (const float*)d_in[iS2L2]    },
    };
    const float* fc2W = (const float*)d_in[iFC2];

    // ---- workspace carve-up (all offsets 256B aligned) ----
    char* ws = (char*)d_ws;
    size_t off = 0;
    auto carve = [&](size_t bytes) -> char* {
        char* p = ws + off;
        off = (off + bytes + 255) & ~(size_t)255;
        return p;
    };
    float*    new_xyz = (float*)   carve((size_t)B_ * S_ * 3 * 4);
    float*    coef    = (float*)   carve(2 * 256 * 4);
    float*    wpad    = (float*)   carve(64 * 32 * 4);
    _Float16* g       = (_Float16*)carve((size_t)B_ * S_ * 64 * 32 * 2);      // 16MB
    _Float16* actA    = (_Float16*)carve((size_t)B_ * S_ * 64 * 128 * 2);     // 64MB
    _Float16* actB    = (_Float16*)carve((size_t)B_ * S_ * 64 * 128 * 2);     // 64MB
    _Float16* fc1in   = (_Float16*)carve((size_t)B_ * 4608 * 2);
    _Float16* fcb1    = (_Float16*)carve((size_t)B_ * 64 * 2);
    _Float16* fcb2    = (_Float16*)carve((size_t)B_ * 256 * 2);

    // NT chosen by N; grid = (Mtiles/8, Nstrips). Mtiles is always a multiple
    // of 8 (4096/8192/16384 conv, 16 FC).
    auto launch_gemm = [&](const _Float16* A, const float* W, _Float16* O,
                           int M, int N, int K) {
        const int mb = (M / 16) / 8;
        if (N == 32)
            gemm_wmma_nt<2><<<dim3(mb, 1), 256, 0, stream>>>(A, W, O, M, N, K);
        else if (N == 64)
            gemm_wmma_nt<4><<<dim3(mb, 1), 256, 0, stream>>>(A, W, O, M, N, K);
        else
            gemm_wmma_nt<8><<<dim3(mb, N / 128), 256, 0, stream>>>(A, W, O, M, N, K);
    };

    // ---- 1) FPS centers ----
    fps_kernel<<<B_, 256, 0, stream>>>(xyz, new_xyz);

    // ---- 2) three MSG scales ----
    const float radii2[3]  = {0.01f, 0.04f, 0.16f};
    const int   Ksamp[3]   = {16, 32, 64};
    const int   Couts[3][3] = {{32, 32, 32}, {32, 64, 128}, {64, 128, 128}};
    const int   chanOff[3] = {0, 32, 160};
    for (int sc = 0; sc < 3; ++sc) {
        const int K = Ksamp[sc];
        const int M = B_ * S_ * K;
        ballgroup_kernel<<<B_ * S_, 256, 0, stream>>>(xyz, new_xyz, g, radii2[sc], K);
        const _Float16* cur = g;
        int Kin = 32;                              // grouped input zero-padded 6->32
        for (int L = 0; L < 3; ++L) {
            const int Cout = Couts[sc][L];
            _Float16* outp = (L & 1) ? actB : actA;
            const float* Wl = Ws[sc][L];
            if (L == 0) {                          // pad (Cout,6) -> (Cout,32)
                padw_kernel<<<(Cout * 32 + 255) / 256, 256, 0, stream>>>(Wl, wpad, Cout);
                Wl = wpad;
            }
            launch_gemm(cur, Wl, outp, M, Cout, Kin);
            stats_kernel<<<Cout, 256, 0, stream>>>(outp, coef, M, Cout);
            const size_t tot = (size_t)M * Cout;
            bnrelu_kernel<<<(int)((tot + 255) / 256), 256, 0, stream>>>(outp, coef, M, Cout, nullptr);
            cur = outp;
            Kin = Cout;
        }
        const int Cl = Couts[sc][2];
        pool_kernel<<<(B_ * S_ * Cl + 255) / 256, 256, 0, stream>>>(cur, fc1in, K, Cl, chanOff[sc]);
    }

    // ---- 3) FC1 [256,4608]x[64,4608]^T -> BN -> ReLU ----
    {
        const int M = B_, N = 64, K = 4608;
        launch_gemm(fc1in, fc1W, fcb1, M, N, K);
        stats_kernel<<<N, 256, 0, stream>>>(fcb1, coef, M, N);
        bnrelu_kernel<<<(M * N + 255) / 256, 256, 0, stream>>>(fcb1, coef, M, N, nullptr);
    }
    // ---- 4) FC2 [256,64]x[256,64]^T -> BN -> ReLU -> d_out (f32) ----
    {
        const int M = B_, N = 256, K = 64;
        launch_gemm(fcb1, fc2W, fcb2, M, N, K);
        stats_kernel<<<N, 256, 0, stream>>>(fcb2, coef, M, N);
        bnrelu_kernel<<<(M * N + 255) / 256, 256, 0, stream>>>(fcb2, coef, M, N, (float*)d_out);
    }
}
